// MultiHeadAttention_28226525070343
// MI455X (gfx1250) — matrix operand order repaired and checked by reading, not compiled, not run on hardware
//
#include <hip/hip_runtime.h>
#include <hip/hip_bf16.h>
#include <cstdint>
#include <cstddef>

// ---------------------------------------------------------------------------
// MI455X (gfx1250) multi-head attention, bf16 WMMA everywhere.
//   D_MODEL=1024, H=16, Dk=64, B=2, S=2048  ->  M = B*S = 4096
// Matrix-core bound (~68 GFLOP vs ~100 MB HBM): bf16 WMMA + f32 accumulate,
// double-buffered LDS, async global->LDS copies, TDM tensor loads for K tiles.
// ---------------------------------------------------------------------------

typedef __bf16 bf16_t;
typedef __attribute__((ext_vector_type(16))) __bf16 v16bf;
typedef __attribute__((ext_vector_type(8)))  __bf16 v8bf;
typedef __attribute__((ext_vector_type(8)))  float  v8f;
typedef __attribute__((ext_vector_type(4)))  int    v4i;

#define D_MODEL   1024
#define NUM_HEADS 16
#define D_K       64
#define BATCH     2
#define SEQ       2048
#define M_TOTAL   (BATCH * SEQ)   // 4096

// --- async global->LDS (CDNA5 GLOBAL_LOAD_ASYNC_TO_LDS_B128, ASYNCcnt) ---
#if __has_builtin(__builtin_amdgcn_global_load_async_to_lds_b128) && \
    __has_builtin(__builtin_amdgcn_s_wait_asynccnt)
#define HAVE_ASYNC_LDS 1
typedef __attribute__((address_space(1))) v4i g_v4i;
typedef __attribute__((address_space(3))) v4i l_v4i;
#endif

// --- TDM (CDNA5 TENSOR_LOAD_TO_LDS, TENSORcnt) ---
#if __has_builtin(__builtin_amdgcn_tensor_load_to_lds) && \
    __has_builtin(__builtin_amdgcn_s_wait_tensorcnt)
#define HAVE_TDM 1
typedef __attribute__((ext_vector_type(4))) unsigned u32x4;
typedef __attribute__((ext_vector_type(8))) int      i32x8;
typedef __attribute__((ext_vector_type(4))) int      i32x4;
#endif
typedef __attribute__((address_space(3))) bf16_t lds_bf16;

__device__ __forceinline__ void copy16_g2l(const bf16_t* g, bf16_t* l) {
#ifdef HAVE_ASYNC_LDS
  __builtin_amdgcn_global_load_async_to_lds_b128((g_v4i*)(g), (l_v4i*)(l), 0,
                                                 0);
#else
  *(v8bf*)l = *(const v8bf*)g;
#endif
}

__device__ __forceinline__ void wait_async() {
#ifdef HAVE_ASYNC_LDS
  __builtin_amdgcn_s_wait_asynccnt(0);
#endif
}

__device__ __forceinline__ void wait_tensor() {
#ifdef HAVE_TDM
  __builtin_amdgcn_s_wait_tensorcnt(0);
#endif
}

#ifdef HAVE_TDM
// One-shot TDM load of a rows x 64 bf16 row-major tile (contiguous rows of a
// [SEQ x 64] tensor) into LDS.  D# layout per CDNA5 ISA ch.8.
__device__ __forceinline__ void tdm_load_tile64(const bf16_t* g, bf16_t* l,
                                                int rows) {
  const unsigned long long ga = (unsigned long long)(size_t)g;
  u32x4 g0;
  g0[0] = 1u;                                   // count=1 (valid), user mode
  g0[1] = (unsigned)(size_t)(lds_bf16*)l;       // lds_addr (bytes)
  g0[2] = (unsigned)ga;                         // global_addr[31:0]
  g0[3] = (unsigned)((ga >> 32) & 0x01FFFFFFull) | (2u << 30);  // addr|type=2
  i32x8 g1;
  g1[0] = 1 << 16;                              // data_size=1 -> 2 bytes
  g1[1] = 64 << 16;                             // tensor_dim0 = 64
  g1[2] = (int)(2048u << 16);                   // tensor_dim1 = 2048 (lo16)
  g1[3] = 64 << 16;                             // tile_dim0 = 64
  g1[4] = rows;                                 // tile_dim1 = rows
  g1[5] = 64;                                   // tensor_dim0_stride = 64
  g1[6] = 0;
  g1[7] = 0;
  i32x4 z4;
  z4[0] = z4[1] = z4[2] = z4[3] = 0;            // groups 2/3 unused (2D tile)
  i32x8 z8;
#pragma unroll
  for (int i = 0; i < 8; ++i) z8[i] = 0;
  __builtin_amdgcn_tensor_load_to_lds(g0, g1, z4, z4, z8, 0);
}
#endif

// round-to-nearest-even f32 -> bf16 (scalar epilogue stores)
__device__ __forceinline__ bf16_t f2bf(float f) {
  unsigned u = __builtin_bit_cast(unsigned, f);
  unsigned r = u + 0x7FFFu + ((u >> 16) & 1u);
  unsigned short h = (unsigned short)(r >> 16);
  return __builtin_bit_cast(bf16_t, h);
}

// pack two f32 -> bf16 pair: 2 adds + 1 v_perm_b32 (round-to-nearest)
__device__ __forceinline__ unsigned pk2bf(float a, float b) {
  unsigned ua = __builtin_bit_cast(unsigned, a) + 0x8000u;
  unsigned ub = __builtin_bit_cast(unsigned, b) + 0x8000u;
  // perm(S0=ua, S1=ub): result bytes = sel picks {0-3: ub, 4-7: ua}
  // 0x07060302 -> [ua.b3 ua.b2 ub.b3 ub.b2]?  we want lo16=bf(a), hi16=bf(b)
  return __builtin_amdgcn_perm(ua, ub, 0x07060302u);
}

__device__ __forceinline__ v8f v8f_zero() {
  v8f z;
#pragma unroll
  for (int i = 0; i < 8; ++i) z[i] = 0.0f;
  return z;
}

__device__ __forceinline__ v16bf mk16(v8bf lo, v8bf hi) {
  v16bf r;
#pragma unroll
  for (int i = 0; i < 8; ++i) { r[i] = lo[i]; r[i + 8] = hi[i]; }
  return r;
}

// convert 16 staged f32 -> bf16 and store 32B to LDS (2x ds_store_b128)
__device__ __forceinline__ void cvt16_store(const float4* f, bf16_t* dst) {
  uint4 lo, hi;
  lo.x = pk2bf(f[0].x, f[0].y); lo.y = pk2bf(f[0].z, f[0].w);
  lo.z = pk2bf(f[1].x, f[1].y); lo.w = pk2bf(f[1].z, f[1].w);
  hi.x = pk2bf(f[2].x, f[2].y); hi.y = pk2bf(f[2].z, f[2].w);
  hi.z = pk2bf(f[3].x, f[3].y); hi.w = pk2bf(f[3].z, f[3].w);
  ((uint4*)dst)[0] = lo;
  ((uint4*)dst)[1] = hi;
}

// ---------------------------------------------------------------------------
// GEMM: Y = X @ W^T + bias.  X: [M x 1024] (f32 or bf16), W: [1024 x 1024]
// f32 row-major.  MODE 0: bf16 scattered to [B,H,S,Dk]; MODE 1: f32 row-major.
// 256 threads = 8 waves (4M x 2N); WG tile 128x128, wave tile 32x64,
// K-step 32, double-buffered LDS.
// ---------------------------------------------------------------------------
template <bool A_BF16, int MODE>
__global__ __launch_bounds__(256)
void mha_gemm_kernel(const void* __restrict__ Xv, const float* __restrict__ W,
                     const float* __restrict__ bias, void* __restrict__ Yv) {
  __shared__ alignas(16) bf16_t As[2][128][32];
  __shared__ alignas(16) bf16_t Bs[2][128][32];

  const int tid  = threadIdx.x;
  const int lane = tid & 31;
  const int wave = tid >> 5;
  const int wm   = wave & 3;
  const int wn   = wave >> 2;
  const int half = lane >> 4;
  const int l16  = lane & 15;

  const int m0 = blockIdx.x * 128;
  const int n0 = blockIdx.y * 128;

  const int row = tid >> 1;            // 0..127
  const int kc  = (tid & 1) * 16;      // 0 or 16

  v8f acc[2][4];
#pragma unroll
  for (int mt = 0; mt < 2; ++mt)
#pragma unroll
    for (int nt = 0; nt < 4; ++nt) acc[mt][nt] = v8f_zero();

  // ---- prologue: tile 0 into buffer 0 ----
  {
    if (A_BF16) {
      const bf16_t* src = (const bf16_t*)Xv + (size_t)(m0 + row) * D_MODEL + kc;
      copy16_g2l(src,     &As[0][row][kc]);
      copy16_g2l(src + 8, &As[0][row][kc + 8]);
    } else {
      const float* src = (const float*)Xv + (size_t)(m0 + row) * D_MODEL + kc;
      float4 af[4];
#pragma unroll
      for (int i = 0; i < 4; ++i) af[i] = ((const float4*)src)[i];
      cvt16_store(af, &As[0][row][kc]);
    }
    const float* wsrc = W + (size_t)(n0 + row) * D_MODEL + kc;
    float4 bfr[4];
#pragma unroll
    for (int i = 0; i < 4; ++i) bfr[i] = ((const float4*)wsrc)[i];
    cvt16_store(bfr, &Bs[0][row][kc]);
  }
  wait_async();
  __syncthreads();

  // ---- main loop: compute buf[cur] while loading tile it+1 into buf[nxt] ----
  for (int it = 0; it < D_MODEL / 32; ++it) {
    const int cur = it & 1, nxt = cur ^ 1;
    const int k1 = (it + 1) * 32;
    const bool has_next = (it + 1) < (D_MODEL / 32);

    float4 af[4], bfr[4];
    if (has_next) {
      if (A_BF16) {
        const bf16_t* src =
            (const bf16_t*)Xv + (size_t)(m0 + row) * D_MODEL + k1 + kc;
        copy16_g2l(src,     &As[nxt][row][kc]);
        copy16_g2l(src + 8, &As[nxt][row][kc + 8]);
      } else {
        const float* src =
            (const float*)Xv + (size_t)(m0 + row) * D_MODEL + k1 + kc;
#pragma unroll
        for (int i = 0; i < 4; ++i) af[i] = ((const float4*)src)[i];
      }
      const float* wsrc = W + (size_t)(n0 + row) * D_MODEL + k1 + kc;
#pragma unroll
      for (int i = 0; i < 4; ++i) bfr[i] = ((const float4*)wsrc)[i];
    }

    // fragments (CDNA5 16-bit A 16x32 / B 32x16 VGPR layouts)
    v16bf afrag[2], bfrag[4];
#pragma unroll
    for (int mt = 0; mt < 2; ++mt) {
      const int r = wm * 32 + mt * 16 + l16;
      v8bf lo = *(const v8bf*)&As[cur][r][half * 8];
      v8bf hi = *(const v8bf*)&As[cur][r][16 + half * 8];
      afrag[mt] = mk16(lo, hi);
    }
#pragma unroll
    for (int nt = 0; nt < 4; ++nt) {
      const int c = wn * 64 + nt * 16 + l16;
      v8bf lo = *(const v8bf*)&Bs[cur][c][half * 8];
      v8bf hi = *(const v8bf*)&Bs[cur][c][16 + half * 8];
      bfrag[nt] = mk16(lo, hi);
    }
#pragma unroll
    for (int mt = 0; mt < 2; ++mt)
#pragma unroll
      for (int nt = 0; nt < 4; ++nt)
        acc[mt][nt] = __builtin_amdgcn_wmma_f32_16x16x32_bf16(
            false, afrag[mt], false, bfrag[nt], (short)0, acc[mt][nt],
            false, false);

    if (has_next) {
      if (!A_BF16) cvt16_store(af, &As[nxt][row][kc]);
      cvt16_store(bfr, &Bs[nxt][row][kc]);
    }
    wait_async();
    __syncthreads();
  }

  // ---- epilogue: bias + store (C layout: row = half*8 + v, col = l16) ----
#pragma unroll
  for (int nt = 0; nt < 4; ++nt) {
    const int n   = n0 + wn * 64 + nt * 16 + l16;
    const float bv = bias[n];
#pragma unroll
    for (int mt = 0; mt < 2; ++mt) {
#pragma unroll
      for (int v = 0; v < 8; ++v) {
        const int m = m0 + wm * 32 + mt * 16 + half * 8 + v;
        const float val = acc[mt][nt][v] + bv;
        if (MODE == 0) {
          const int b = m >> 11, s = m & (SEQ - 1);
          const int h = n >> 6, dk = n & 63;
          ((bf16_t*)Yv)[((((size_t)b * NUM_HEADS + h) * SEQ + s) << 6) + dk] =
              f2bf(val);
        } else {
          ((float*)Yv)[(size_t)m * D_MODEL + n] = val;
        }
      }
    }
  }
}

// ---------------------------------------------------------------------------
// Flash attention over one (b,h): 128 threads (4 waves), each wave owns 16
// query rows; 32-key K/V tiles stream through double-buffered LDS.
// K tiles: one TDM tensor load per tile (wave 0), else async global->LDS.
// ---------------------------------------------------------------------------
__global__ __launch_bounds__(128)
void mha_flash_attn_kernel(const bf16_t* __restrict__ qw,
                           const bf16_t* __restrict__ kw,
                           const bf16_t* __restrict__ vw,
                           bf16_t* __restrict__ ow) {
  __shared__ alignas(16) bf16_t Ks[2][32][64];   // keys x dims
  __shared__ alignas(16) bf16_t Vt[2][64][32];   // dims x keys (transposed)
  __shared__ alignas(16) bf16_t Ps[4][16][32];   // per-wave P re-striping

  const int tid  = threadIdx.x;
  const int lane = tid & 31;
  const int wave = tid >> 5;
  const int half = lane >> 4;
  const int l16  = lane & 15;

  const int bh = blockIdx.y;                       // b*16 + h
  const int q0 = blockIdx.x * 64 + wave * 16;
  const size_t base = (size_t)bh * SEQ * D_K;

  const int kr = tid >> 2;          // key row 0..31 (loader mapping)
  const int d0 = (tid & 3) * 16;    // dim chunk

  // load the wave's q fragments (A-matrix layout, rows = l16)
  v16bf qf[2];
  {
    const bf16_t* qrow = qw + base + (size_t)(q0 + l16) * D_K;
#pragma unroll
    for (int f = 0; f < 2; ++f) {
      v8bf lo = *(const v8bf*)(qrow + f * 32 + half * 8);
      v8bf hi = *(const v8bf*)(qrow + f * 32 + 16 + half * 8);
      qf[f] = mk16(lo, hi);
    }
  }

  v8f oacc[4];
#pragma unroll
  for (int dn = 0; dn < 4; ++dn) oacc[dn] = v8f_zero();
  float mrow[8], lrow[8];
#pragma unroll
  for (int v = 0; v < 8; ++v) { mrow[v] = -3.0e38f; lrow[v] = 0.0f; }

  const float sc = 0.125f * 1.44269504088896340736f;  // (1/sqrt(64))*log2(e)

  // tile issue: K via TDM (wave 0) or per-lane async; V staged + transposed
  auto issue_tile = [&](int kk0, int buf) {
#ifdef HAVE_TDM
    if (wave == 0)
      tdm_load_tile64(kw + base + (size_t)kk0 * D_K, &Ks[buf][0][0], 32);
#else
    const bf16_t* ksrc = kw + base + (size_t)(kk0 + kr) * D_K + d0;
    copy16_g2l(ksrc,     &Ks[buf][kr][d0]);
    copy16_g2l(ksrc + 8, &Ks[buf][kr][d0 + 8]);
#endif
    const bf16_t* vsrc = vw + base + (size_t)(kk0 + kr) * D_K + d0;
    v8bf v0 = ((const v8bf*)vsrc)[0];
    v8bf v1 = ((const v8bf*)vsrc)[1];
#pragma unroll
    for (int j = 0; j < 8; ++j) {
      Vt[buf][d0 + j][kr]     = v0[j];
      Vt[buf][d0 + 8 + j][kr] = v1[j];
    }
  };

  issue_tile(0, 0);

  for (int it = 0; it < SEQ / 32; ++it) {
    const int cur = it & 1;
    wait_async();
    wait_tensor();
    __syncthreads();                       // buf[cur] ready for everyone
    if (it + 1 < SEQ / 32) issue_tile((it + 1) * 32, cur ^ 1);

    // ---- scores: load all 4 B-fragments first, then 4 WMMAs ----
    v16bf bk2[2][2];
#pragma unroll
    for (int kn = 0; kn < 2; ++kn)
#pragma unroll
      for (int ds = 0; ds < 2; ++ds) {
        const bf16_t* kp = &Ks[cur][kn * 16 + l16][ds * 32 + half * 8];
        bk2[kn][ds] = mk16(*(const v8bf*)kp, *(const v8bf*)(kp + 16));
      }
    v8f sacc[2];
#pragma unroll
    for (int kn = 0; kn < 2; ++kn) {
      sacc[kn] = __builtin_amdgcn_wmma_f32_16x16x32_bf16(
          false, qf[0], false, bk2[kn][0], (short)0, v8f_zero(), false, false);
      sacc[kn] = __builtin_amdgcn_wmma_f32_16x16x32_bf16(
          false, qf[1], false, bk2[kn][1], (short)0, sacc[kn], false, false);
    }

    // ---- online softmax (row = half*8 + v; 16-lane row groups) ----
#pragma unroll
    for (int v = 0; v < 8; ++v) {
      float a0 = sacc[0][v] * sc;
      float a1 = sacc[1][v] * sc;
      float mx = fmaxf(a0, a1);
#pragma unroll
      for (int off = 1; off < 16; off <<= 1)
        mx = fmaxf(mx, __shfl_xor(mx, off, 16));
      const float mnew  = fmaxf(mrow[v], mx);
      const float alpha = exp2f(mrow[v] - mnew);
      const float p0 = exp2f(a0 - mnew);
      const float p1 = exp2f(a1 - mnew);
      float ps = p0 + p1;
#pragma unroll
      for (int off = 1; off < 16; off <<= 1)
        ps += __shfl_xor(ps, off, 16);
      lrow[v] = lrow[v] * alpha + ps;
      mrow[v] = mnew;
#pragma unroll
      for (int dn = 0; dn < 4; ++dn) oacc[dn][v] *= alpha;
      const int r = half * 8 + v;
      Ps[wave][r][l16]      = f2bf(p0);
      Ps[wave][r][16 + l16] = f2bf(p1);
    }

    // ---- re-stripe P into A-fragment layout (per-wave LDS buffer) ----
    v16bf pf;
    {
      const bf16_t* pr = &Ps[wave][l16][0];
      v8bf lo = *(const v8bf*)(pr + half * 8);
      v8bf hi = *(const v8bf*)(pr + 16 + half * 8);
      pf = mk16(lo, hi);
    }

    // ---- O += P @ V: load all 4 B-fragments, then 4 WMMAs ----
    v16bf bv4[4];
#pragma unroll
    for (int dn = 0; dn < 4; ++dn) {
      const bf16_t* vp = &Vt[cur][dn * 16 + l16][half * 8];
      bv4[dn] = mk16(*(const v8bf*)vp, *(const v8bf*)(vp + 16));
    }
#pragma unroll
    for (int dn = 0; dn < 4; ++dn)
      oacc[dn] = __builtin_amdgcn_wmma_f32_16x16x32_bf16(
          false, pf, false, bv4[dn], (short)0, oacc[dn], false, false);
  }

  // ---- normalize + store to [B, S, 1024] (col = h*64 + dn*16 + l16) ----
  const int b = bh >> 4, h = bh & 15;
#pragma unroll
  for (int v = 0; v < 8; ++v) {
    const float inv = 1.0f / lrow[v];
    const int s = q0 + half * 8 + v;
    const size_t orow = ((size_t)b * SEQ + s) * D_MODEL + (size_t)h * D_K;
#pragma unroll
    for (int dn = 0; dn < 4; ++dn)
      ow[orow + dn * 16 + l16] = f2bf(oacc[dn][v] * inv);
  }
}

// ---------------------------------------------------------------------------
extern "C" void kernel_launch(void* const* d_in, const int* in_sizes, int n_in,
                              void* d_out, int out_size, void* d_ws,
                              size_t ws_size, hipStream_t stream) {
  (void)in_sizes; (void)n_in; (void)out_size; (void)ws_size;
  const float* Q  = (const float*)d_in[0];
  const float* K  = (const float*)d_in[1];
  const float* V  = (const float*)d_in[2];
  const float* Wq = (const float*)d_in[3];
  const float* bq = (const float*)d_in[4];
  const float* Wk = (const float*)d_in[5];
  const float* bk = (const float*)d_in[6];
  const float* Wv = (const float*)d_in[7];
  const float* bv = (const float*)d_in[8];
  const float* Wo = (const float*)d_in[9];
  const float* bo = (const float*)d_in[10];

  // workspace: bf16 q, k, v in [B,H,S,64] + bf16 attention output [B,S,1024]
  const size_t elems = (size_t)M_TOTAL * D_MODEL;  // 4 Mi elements
  bf16_t* qws = (bf16_t*)d_ws;
  bf16_t* kws = qws + elems;
  bf16_t* vws = kws + elems;
  bf16_t* aws = vws + elems;

  const dim3 ggrid(M_TOTAL / 128, D_MODEL / 128);  // 32 x 8
  mha_gemm_kernel<false, 0><<<ggrid, 256, 0, stream>>>(Q, Wq, bq, qws);
  mha_gemm_kernel<false, 0><<<ggrid, 256, 0, stream>>>(K, Wk, bk, kws);
  mha_gemm_kernel<false, 0><<<ggrid, 256, 0, stream>>>(V, Wv, bv, vws);

  mha_flash_attn_kernel<<<dim3(SEQ / 64, BATCH * NUM_HEADS), 128, 0, stream>>>(
      qws, kws, vws, aws);

  mha_gemm_kernel<true, 1><<<ggrid, 256, 0, stream>>>(aws, Wo, bo,
                                                      (float*)d_out);
}
